// ESN_80281528697360
// MI455X (gfx1250) — matrix operand order repaired and checked by reading, not compiled, not run on hardware
//
#include <hip/hip_runtime.h>
#include <cstdint>
#include <cstddef>

// ---------------- Problem constants ----------------
#define BATCH   8
#define TLEN    4096
#define D_IN    64
#define D_RES   1024

// ---------------- Kernel geometry ------------------
#define NWG         16              // one CDNA5 cluster (max 16 WGs), 1 WG per WGP
#define ROWS_PER_WG (D_RES / NWG)   // 64 reservoir rows per WG
#define KTOT        (D_RES + D_IN)  // 1088: fused [W | W_in] contraction dim
#define THREADS     128             // 4 wave32 -> 4 M-tiles of 16 rows
// LDS row/col stride in bf16 halves: 1096 halves = 548 dwords.
// gcd(548,64)=4 with full period 16 -> 16 lanes with distinct rows/cols hit
// 16 distinct LDS banks (conflict-free fragment feeds).
#define STRIDE_H    1096
#define A_BYTES     (ROWS_PER_WG * STRIDE_H * 2)   // 140,288 B : [W|Win] slice, bf16
#define HB_BYTES    (16 * STRIDE_H * 2)            //  35,072 B : B operand [16 cols][K]
#define BI_BYTES    (ROWS_PER_WG * 4)              //     256 B : bias slice
#define SMEM_BYTES  (A_BYTES + HB_BYTES + BI_BYTES)

// ---------------- Workspace layout (bytes) ----------
#define CNT_OFF   0u                         // barrier counter + phase (256 B)
#define H_OFF     256u                       // hshared[2][8][1024] bf16 = 32 KB
#define W_OFF     (H_OFF + 2u*BATCH*D_RES*2u)            // W bf16, 2 MB
#define WIN_OFF   (W_OFF + (unsigned)D_RES*D_RES*2u)     // W_in bf16, 128 KB
#define U_OFF     (WIN_OFF + (unsigned)D_RES*D_IN*2u)    // u staged [t][b][64] bf16, 4 MB

typedef __attribute__((ext_vector_type(16))) __bf16       v16bf;
typedef __attribute__((ext_vector_type(8)))  float        v8f;
typedef __attribute__((ext_vector_type(4)))  float        f32x4;
typedef __attribute__((ext_vector_type(4)))  unsigned int u32x4;

// 16-byte async global->LDS copy (ASYNCcnt path). LDS address = low 32 bits of
// the generic pointer (generic LDS addr = SHARED_BASE:offset).
__device__ __forceinline__ void async_copy16(const void* g, const void* l) {
  unsigned           lo = (unsigned)(unsigned long long)(uintptr_t)l;
  unsigned long long ga = (unsigned long long)(uintptr_t)g;
  asm volatile("global_load_async_to_lds_b128 %0, %1, off"
               :: "v"(lo), "v"(ga) : "memory");
}
__device__ __forceinline__ void wait_async0() {
  asm volatile("s_wait_asynccnt 0x0" ::: "memory");
}

// Branch-free tanh on the v_exp_f32 / v_rcp_f32 transcendental path.
// tanh(x) = 1 - 2/(1+e^{2x}); e^{2x} = exp2(x * 2*log2(e)).
// x -> +inf: exp2 -> inf, rcp(inf)=0 -> 1.   x -> -inf: exp2 -> 0, rcp(1)=1 -> -1.
__device__ __forceinline__ float fast_tanh(float x) {
  float e = __builtin_amdgcn_exp2f(x * 2.885390081777926814720f);
  return 1.0f - 2.0f * __builtin_amdgcn_rcpf(e + 1.0f);
}

// ---------------------------------------------------------------------------
// Prep: convert W / W_in / u to bf16 staging buffers, zero h double-buffer and
// barrier state. Runs every launch (deterministic, graph-capture safe).
// ---------------------------------------------------------------------------
__global__ void esn_prep(const float* __restrict__ w,
                         const float* __restrict__ w_in,
                         const float* __restrict__ u,
                         unsigned char* __restrict__ ws)
{
  const size_t i = (size_t)blockIdx.x * blockDim.x + threadIdx.x;
  __bf16* hsh  = (__bf16*)(ws + H_OFF);
  __bf16* wb   = (__bf16*)(ws + W_OFF);
  __bf16* winb = (__bf16*)(ws + WIN_OFF);
  __bf16* ub   = (__bf16*)(ws + U_OFF);

  if (i < 64) ((int*)(ws + CNT_OFF))[i] = 0;                 // barrier cnt/phase
  if (i < 2u * BATCH * D_RES) hsh[i] = (__bf16)0.0f;          // h0 = 0 (both bufs)
  if (i < (size_t)D_RES * D_RES) wb[i] = (__bf16)w[i];
  if (i < (size_t)D_RES * D_IN)  winb[i] = (__bf16)w_in[i];
  if (i < (size_t)BATCH * TLEN * D_IN) {                      // u[b][t][k] -> ub[t][b][k]
    int    k = (int)(i % D_IN);
    size_t r = i / D_IN;
    int    t = (int)(r % TLEN);
    int    b = (int)(r / TLEN);
    ub[((size_t)t * BATCH + b) * D_IN + k] = (__bf16)u[i];
  }
}

// ---------------------------------------------------------------------------
// Persistent scan kernel: 16 WGs (one cluster) x 128 threads.
// WG g owns reservoir rows [g*64, g*64+64). A = [W | W_in] slice lives in LDS
// for the whole run. Each step: async-stage [h ; u_t] (bf16) into LDS, do the
// 64x16(K=1088) GEMM via v_wmma_f32_16x16x32_bf16, tanh+bias, publish h in
// bf16 to a double-buffered L2-resident broadcast buffer, then cluster-sync.
// ---------------------------------------------------------------------------
__launch_bounds__(THREADS, 1)
__global__ void esn_scan(const float* __restrict__ w_bias,
                         unsigned char* __restrict__ ws,
                         float* __restrict__ out)
{
  extern __shared__ char smem[];
  __bf16* A  = (__bf16*)smem;                       // [64][STRIDE_H]
  __bf16* HB = (__bf16*)(smem + A_BYTES);           // [16][STRIDE_H]  (B operand)
  float*  BI = (float*)(smem + A_BYTES + HB_BYTES); // [64] bias slice

  const __bf16* Wb   = (const __bf16*)(ws + W_OFF);
  const __bf16* Winb = (const __bf16*)(ws + WIN_OFF);
  const __bf16* Ub   = (const __bf16*)(ws + U_OFF);
  __bf16*       Hsh  = (__bf16*)(ws + H_OFF);       // [2][8][1024]
  int* bcnt   = (int*)(ws + CNT_OFF);
  int* bphase = bcnt + 1;

  const int tid = threadIdx.x;
  const int g   = blockIdx.x;
  const int r0  = g * ROWS_PER_WG;

  // ---- one-time setup: zero B pad columns, load [W | W_in] slice, bias ----
  for (int i = tid; i < 16 * STRIDE_H; i += THREADS) HB[i] = (__bf16)0.0f;
  for (int i = tid; i < ROWS_PER_WG * KTOT; i += THREADS) {
    int row = i / KTOT, k = i - row * KTOT;
    __bf16 v = (k < D_RES) ? Wb[(size_t)(r0 + row) * D_RES + k]
                           : Winb[(r0 + row) * D_IN + (k - D_RES)];
    A[row * STRIDE_H + k] = v;
  }
  if (tid < ROWS_PER_WG) BI[tid] = w_bias[r0 + tid];
  __syncthreads();

  // ---- per-lane WMMA fragment geometry (contiguous k = 16h + s, same for A and B) ----
  const int wv  = tid >> 5;                 // wave -> M-tile (16 rows)
  const int ln  = tid & 31;
  const int aRow = (wv << 4) + (ln & 15);   // A: lanes 0-15 M=lane, 16-31 M=lane-16
  const int aKb  = (ln < 16) ? 0 : 16;      // A: lo lanes K 0-15, hi K 16-31
  const int bN   = ln & 15;                 // B: lane -> column N
  const int bKb  = (ln < 16) ? 0 : 16;      // B: lo lanes K 0-15, hi K 16-31
  const __bf16* Arow = A  + aRow * STRIDE_H;
  const __bf16* Bcol = HB + bN   * STRIDE_H;

  // Epilogue geometry + loop-invariant bias preload (registers, not LDS).
  const int col   = ln & 15;            // N = batch lane (8..15 are pad)
  const int mr0   = (ln >> 4) << 3;     // D rows: VGPR j -> M = j (+8 hi lanes)
  const int rloc0 = (wv << 4) + mr0;    // local row of acc[0] (8 consecutive)
  float bias_r[8];
#pragma unroll
  for (int j = 0; j < 8; ++j) bias_r[j] = BI[rloc0 + j];

  int myph = 0;

  for (int t = 0; t < TLEN; ++t) {
    const __bf16* hr = Hsh + (size_t)((t & 1) ^ 1) * (BATCH * D_RES); // read h_{t-1}
    __bf16*       hw = Hsh + (size_t)(t & 1)       * (BATCH * D_RES); // write h_t

    // ---- async stage h_{t-1} (16 KB) into HB cols 0..7, k in [0,1024) ----
    for (int c = tid; c < (BATCH * D_RES * 2) / 16; c += THREADS) {   // 1024 chunks
      int n = c >> 7;                    // 128 x 16B chunks per column
      int off = (c & 127) << 4;
      async_copy16((const char*)hr + n * (D_RES * 2) + off,
                   (const char*)HB + n * (STRIDE_H * 2) + off);
    }
    // ---- async stage u_t (1 KB) into HB cols 0..7, k in [1024,1088) ----
    if (tid < 64) {
      int n = tid >> 3, off = (tid & 7) << 4;
      async_copy16((const char*)(Ub + ((size_t)t * BATCH + n) * D_IN) + off,
                   (const char*)HB + n * (STRIDE_H * 2) + D_RES * 2 + off);
    }
    wait_async0();
    __syncthreads();

    // ---- 16x16 (K=1088) tile: fully unrolled 34 x v_wmma_f32_16x16x32_bf16 ----
    v8f acc = {0.f, 0.f, 0.f, 0.f, 0.f, 0.f, 0.f, 0.f};
#pragma unroll
    for (int k0 = 0; k0 < KTOT; k0 += 32) {
      union { u32x4 q[2]; v16bf v; } ua;
      ua.q[0] = *(const u32x4*)(Arow + k0 + aKb);
      ua.q[1] = *(const u32x4*)(Arow + k0 + aKb + 8);
      v16bf bf = *(const v16bf*)(Bcol + k0 + bKb);
      acc = __builtin_amdgcn_wmma_f32_16x16x32_bf16(
          false, ua.v, false, bf, (short)0, acc, false, false);
    }

    // ---- epilogue: bias + fast tanh; fp32 -> d_out, bf16 -> broadcast buf ----
    {
      float xs[8];
#pragma unroll
      for (int j = 0; j < 8; ++j) xs[j] = fast_tanh(acc[j] + bias_r[j]);
      if (col < BATCH) {
        const int r = r0 + rloc0;
        float* dst = out + (((size_t)col * TLEN) + t) * D_RES + r;
        f32x4 o0 = {xs[0], xs[1], xs[2], xs[3]};
        f32x4 o1 = {xs[4], xs[5], xs[6], xs[7]};
        *(f32x4*)dst       = o0;
        *(f32x4*)(dst + 4) = o1;
        unsigned hp[4];
#pragma unroll
        for (int j = 0; j < 4; ++j) {
          unsigned lo16 = __builtin_bit_cast(unsigned short, (__bf16)xs[2 * j]);
          unsigned hi16 = __builtin_bit_cast(unsigned short, (__bf16)xs[2 * j + 1]);
          hp[j] = lo16 | (hi16 << 16);
        }
        u32x4 hv = {hp[0], hp[1], hp[2], hp[3]};
        *(u32x4*)((char*)hw + col * (D_RES * 2) + r * 2) = hv;
      }
    }

    // ---- cluster-wide sync: release h_t, then barrier ----
    __threadfence();
    __syncthreads();
    if (tid == 0) {
      if (atomicAdd(bcnt, 1) == NWG - 1) {
        atomicExch(bcnt, 0);
        __threadfence();
        atomicAdd(bphase, 1);
      } else {
        while (__hip_atomic_load(bphase, __ATOMIC_RELAXED,
                                 __HIP_MEMORY_SCOPE_AGENT) <= myph) {
          __builtin_amdgcn_s_sleep(2);
        }
      }
    }
    myph++;
    // Hardware cluster barrier (one wave signals/waits per WG); NOP if the
    // dispatch is not clustered — the atomic barrier above is authoritative.
    if (tid < 32) __builtin_amdgcn_s_cluster_barrier();
    __syncthreads();
    __threadfence();
  }
}

// ---------------------------------------------------------------------------
extern "C" void kernel_launch(void* const* d_in, const int* in_sizes, int n_in,
                              void* d_out, int out_size, void* d_ws, size_t ws_size,
                              hipStream_t stream) {
  const float* u      = (const float*)d_in[0];   // [8][4096][64]
  const float* w_in   = (const float*)d_in[1];   // [1024][64]
  const float* w      = (const float*)d_in[2];   // [1024][1024]
  const float* w_bias = (const float*)d_in[3];   // [1024]
  float*       out    = (float*)d_out;           // [8][4096][1024]
  unsigned char* ws   = (unsigned char*)d_ws;

  (void)in_sizes; (void)n_in; (void)out_size; (void)ws_size;

  // Allow >64KB dynamic LDS (CDNA5: 320 KB/WGP). Host-side attr set is not a
  // stream op, so it is graph-capture safe and deterministic.
  (void)hipFuncSetAttribute((const void*)esn_scan,
                            hipFuncAttributeMaxDynamicSharedMemorySize,
                            SMEM_BYTES);

  // Stage/convert inputs (2,097,152 work items covers the largest array).
  esn_prep<<<8192, 256, 0, stream>>>(w, w_in, u, ws);

  // Persistent clustered scan: 16 WGs x 128 threads.
  esn_scan<<<NWG, THREADS, SMEM_BYTES, stream>>>(w_bias, ws, out);
}
